// TaylorNet_88046829568375
// MI455X (gfx1250) — matrix operand order repaired and checked by reading, not compiled, not run on hardware
//
#include <hip/hip_runtime.h>
#include <hip/hip_bf16.h>

// ---------------------------------------------------------------------------
// TaylorNet on MI455X (gfx1250).
// Cubic term = GEMM  M[p,b] = sum_k C3[p,k]*rel[b,k]  (65536 x 2048, K=256),
// fused epilogue out3[b] += rel[b,d1(p)]*rel[b,d2(p)]*M[p,b].
// Precision: bf16 hi/lo split, 3 x v_wmma_f32_16x16x32_bf16 per K-chunk.
// Round-4 change: double-buffered A tile (ping-pong in LDS, 203KB total).
// Next tile's global loads are issued before the WMMA section so HBM/L2
// latency hides behind 48 WMMAs; convert+ds_store lands in the other buffer;
// one barrier per iteration. Steady state: 10 ds_load_b128 / 6 WMMA, B-hi
// fragments resident in 64 VGPRs, no spills.
// ---------------------------------------------------------------------------

typedef __attribute__((ext_vector_type(16))) __bf16 v16bf;
typedef __attribute__((ext_vector_type(8)))  float  v8f;

#define DIM      256
#define BTOT     2048
#define TN       128                 // batches per block (blockIdx.y dim)
#define ROWG     64                  // row-groups (blockIdx.x dim)
#define IT_PER_BLK ((DIM * DIM / 32) / ROWG)   // 32 iterations of 32 rows
#define RSTRIDE  264                 // padded LDS row stride (bf16 elems, 16B multiple)
#define ABUF     (32 * RSTRIDE)      // one aHi or aLo plane

union FragU {
  v16bf v;
  uint4 q[2];
};

// ------------------------- kernel 1: c0 + c1 + c2 terms --------------------
__global__ void taylor_loworder(const float* __restrict__ x,
                                const float* __restrict__ offsets,
                                const float* __restrict__ c0,
                                const float* __restrict__ c1,
                                const float* __restrict__ c2,
                                float* __restrict__ partials) {
  __shared__ float relS[8][DIM];
  __shared__ float red[256];
  const int tid = threadIdx.x;
  const int b0  = blockIdx.x * 8;

  for (int e = tid; e < 8 * DIM; e += blockDim.x) {
    const int bb = e >> 8, d = e & 255;
    relS[bb][d] = x[(size_t)(b0 + bb) * DIM + d] - offsets[d];
  }
  __syncthreads();

  float inner[8] = {0.f, 0.f, 0.f, 0.f, 0.f, 0.f, 0.f, 0.f};
  const int i = tid;
  for (int j = 0; j < DIM; ++j) {
    const float w = c2[(size_t)i * DIM + j];
#pragma unroll
    for (int bb = 0; bb < 8; ++bb) inner[bb] += w * relS[bb][j];
  }
  const float c1i = c1[i];
  float contrib[8];
#pragma unroll
  for (int bb = 0; bb < 8; ++bb)
    contrib[bb] = relS[bb][i] * (inner[bb] + c1i);

#pragma unroll 1
  for (int bb = 0; bb < 8; ++bb) {
    red[tid] = contrib[bb];
    __syncthreads();
    for (int s = 128; s > 0; s >>= 1) {
      if (tid < s) red[tid] += red[tid + s];
      __syncthreads();
    }
    if (tid == 0) partials[(size_t)ROWG * BTOT + b0 + bb] = c0[0] + red[0];
    __syncthreads();
  }
}

// convert a prefetched 32-float strip to bf16 hi/lo and store to LDS
__device__ __forceinline__ void cvt_store_tile(__bf16* aHiB, __bf16* aLoB,
                                               int r, int k0, const float4* pf) {
#pragma unroll
  for (int q4 = 0; q4 < 8; ++q4) {
    const float4 f = pf[q4];
    const float vals[4] = {f.x, f.y, f.z, f.w};
#pragma unroll
    for (int j = 0; j < 4; ++j) {
      const float  fv = vals[j];
      const __bf16 hi = (__bf16)fv;
      aHiB[r * RSTRIDE + k0 + q4 * 4 + j] = hi;
      aLoB[r * RSTRIDE + k0 + q4 * 4 + j] = (__bf16)(fv - (float)hi);
    }
  }
}

// ------------------------- kernel 2: cubic term via WMMA -------------------
__global__ void __launch_bounds__(256, 1)
taylor_cubic_wmma(const float* __restrict__ x,
                  const float* __restrict__ offsets,
                  const float* __restrict__ c3,
                  float* __restrict__ partials) {
  extern __shared__ __align__(16) char smem[];
  __bf16* relHi = (__bf16*)smem;                     // [TN][RSTRIDE]
  __bf16* relLo = relHi + TN * RSTRIDE;              // [TN][RSTRIDE]
  __bf16* aBase = relLo + TN * RSTRIDE;              // 2 x (aHi[32][RSTRIDE], aLo[32][RSTRIDE])

  const int tid  = threadIdx.x;
  const int lane = tid & 31;
  const int wave = tid >> 5;
  const int half = lane >> 4;      // wave32 half: selects contiguous K sub-range half*16..half*16+15
  const int l16  = lane & 15;
  const int bB   = blockIdx.y * TN;

  // stage rel tile as bf16 hi/lo split
  for (int e = tid; e < TN * DIM; e += blockDim.x) {
    const int bb = e >> 8, d = e & 255;
    const float r  = x[(size_t)(bB + bb) * DIM + d] - offsets[d];
    const __bf16 hi = (__bf16)r;
    relHi[bb * RSTRIDE + d] = hi;
    relLo[bb * RSTRIDE + d] = (__bf16)(r - (float)hi);
  }
  __syncthreads();

  const int myb = wave * 16 + l16;   // batch (== WMMA output column) for this lane

  // Preload B-hi fragments (8 K-chunks) into registers: 64 VGPRs.
  // B layout (32x16 bf16): lane holds B[half*16 + 0..15, n] = rel[myb, kb..kb+15]
  FragU bregH[8];
#pragma unroll
  for (int kc = 0; kc < 8; ++kc) {
    const int kb = kc * 32 + half * 16;
    const uint4* pbh = (const uint4*)(relHi + myb * RSTRIDE + kb);
    bregH[kc].q[0] = pbh[0]; bregH[kc].q[1] = pbh[1];
  }

  const int   r     = tid >> 3;          // staging row 0..31
  const int   k0    = (tid & 7) * 32;    // staging col base
  const int   pBase = blockIdx.x * (IT_PER_BLK * 32);
  float4 pf[8];

  // prologue: tile 0 -> buffer 0
  {
    const float4* src = (const float4*)(c3 + (size_t)(pBase + r) * DIM + k0);
#pragma unroll
    for (int q4 = 0; q4 < 8; ++q4) pf[q4] = src[q4];
    cvt_store_tile(aBase, aBase + ABUF, r, k0, pf);
  }
  __syncthreads();

  float batchAcc = 0.f;

#pragma unroll 1
  for (int t = 0; t < IT_PER_BLK; ++t) {
    const int p0 = pBase + t * 32;       // global row base of the tile we consume
    __bf16* aHiB = aBase + (t & 1) * (2 * ABUF);
    __bf16* aLoB = aHiB + ABUF;
    const bool hasNext = (t + 1) < IT_PER_BLK;

    // issue next tile's global loads now; they complete behind the WMMAs
    if (hasNext) {
      const float4* src = (const float4*)(c3 + (size_t)(p0 + 32 + r) * DIM + k0);
#pragma unroll
      for (int q4 = 0; q4 < 8; ++q4) pf[q4] = src[q4];
    }

    v8f acc0 = {0.f, 0.f, 0.f, 0.f, 0.f, 0.f, 0.f, 0.f};
    v8f acc1 = {0.f, 0.f, 0.f, 0.f, 0.f, 0.f, 0.f, 0.f};
    FragU aH0, aL0, aH1, aL1, bL;

#pragma unroll
    for (int kc = 0; kc < 8; ++kc) {
      // B-lo frag from LDS (reused by both row tiles)
      const int kb = kc * 32 + half * 16;
      const uint4* pbl = (const uint4*)(relLo + myb * RSTRIDE + kb);
      bL.q[0] = pbl[0]; bL.q[1] = pbl[1];

      // A frags (16x32 bf16): lane holds A[m, half*16+0..15]
      const int ka = kc * 32 + half * 16;
      const uint4* pah0 = (const uint4*)(aHiB + l16 * RSTRIDE + ka);
      aH0.q[0] = pah0[0]; aH0.q[1] = pah0[1];
      const uint4* pal0 = (const uint4*)(aLoB + l16 * RSTRIDE + ka);
      aL0.q[0] = pal0[0]; aL0.q[1] = pal0[1];
      const uint4* pah1 = (const uint4*)(aHiB + (16 + l16) * RSTRIDE + ka);
      aH1.q[0] = pah1[0]; aH1.q[1] = pah1[1];
      const uint4* pal1 = (const uint4*)(aLoB + (16 + l16) * RSTRIDE + ka);
      aL1.q[0] = pal1[0]; aL1.q[1] = pal1[1];

      acc0 = __builtin_amdgcn_wmma_f32_16x16x32_bf16(false, aH0.v, false, bregH[kc].v,
                                                     (short)0, acc0, false, false);
      acc0 = __builtin_amdgcn_wmma_f32_16x16x32_bf16(false, aH0.v, false, bL.v,
                                                     (short)0, acc0, false, false);
      acc0 = __builtin_amdgcn_wmma_f32_16x16x32_bf16(false, aL0.v, false, bregH[kc].v,
                                                     (short)0, acc0, false, false);
      acc1 = __builtin_amdgcn_wmma_f32_16x16x32_bf16(false, aH1.v, false, bregH[kc].v,
                                                     (short)0, acc1, false, false);
      acc1 = __builtin_amdgcn_wmma_f32_16x16x32_bf16(false, aH1.v, false, bL.v,
                                                     (short)0, acc1, false, false);
      acc1 = __builtin_amdgcn_wmma_f32_16x16x32_bf16(false, aL1.v, false, bregH[kc].v,
                                                     (short)0, acc1, false, false);
    }

    // fused epilogue: all 32 rows share d1 = p0>>8; d2 = (p0&255) + row
    const int d1  = p0 >> 8;
    const int d2b = p0 & 255;
    const float relD1 = (float)relHi[myb * RSTRIDE + d1] +
                        (float)relLo[myb * RSTRIDE + d1];
    float s = 0.f;
#pragma unroll
    for (int v = 0; v < 8; ++v) {          // tile0 rows v + 8*half
      const int d2 = d2b + v + half * 8;
      s += acc0[v] * ((float)relHi[myb * RSTRIDE + d2] +
                      (float)relLo[myb * RSTRIDE + d2]);
    }
#pragma unroll
    for (int v = 0; v < 8; ++v) {          // tile1 rows 16 + v + 8*half
      const int d2 = d2b + 16 + v + half * 8;
      s += acc1[v] * ((float)relHi[myb * RSTRIDE + d2] +
                      (float)relLo[myb * RSTRIDE + d2]);
    }
    batchAcc += relD1 * s;

    // convert + store the prefetched tile into the other buffer
    if (hasNext) {
      __bf16* nHi = aBase + ((t + 1) & 1) * (2 * ABUF);
      cvt_store_tile(nHi, nHi + ABUF, r, k0, pf);
    }
    __syncthreads();
  }

  // lanes l and l+16 hold the same output column: combine, then one write
  batchAcc += __shfl_xor(batchAcc, 16, 32);
  if (half == 0)
    partials[(size_t)blockIdx.x * BTOT + bB + myb] = batchAcc;
}

// ------------------------- kernel 3: final reduction -----------------------
__global__ void taylor_reduce(const float* __restrict__ partials,
                              float* __restrict__ out) {
  const int b = blockIdx.x * blockDim.x + threadIdx.x;
  if (b >= BTOT) return;
  float s = partials[(size_t)ROWG * BTOT + b];     // low-order terms slot
  for (int g = 0; g < ROWG; ++g) s += partials[(size_t)g * BTOT + b];
  out[b] = s;
}

extern "C" void kernel_launch(void* const* d_in, const int* in_sizes, int n_in,
                              void* d_out, int out_size, void* d_ws, size_t ws_size,
                              hipStream_t stream) {
  const float* x       = (const float*)d_in[0];
  const float* offsets = (const float*)d_in[1];
  const float* c0      = (const float*)d_in[2];
  const float* c1      = (const float*)d_in[3];
  const float* c2      = (const float*)d_in[4];
  const float* c3      = (const float*)d_in[5];
  float* out      = (float*)d_out;
  float* partials = (float*)d_ws;   // (ROWG+1) * BTOT floats = 532,480 B

  taylor_loworder<<<BTOT / 8, 256, 0, stream>>>(x, offsets, c0, c1, c2, partials);

  dim3 grid(ROWG, BTOT / TN);
  const size_t shmem = (size_t)(2 * TN * RSTRIDE + 4 * ABUF) * sizeof(__bf16);
  taylor_cubic_wmma<<<grid, 256, shmem, stream>>>(x, offsets, c3, partials);

  taylor_reduce<<<(BTOT + 255) / 256, 256, 0, stream>>>(partials, out);
}
